// pretrain_actor_rel_78271484002555
// MI455X (gfx1250) — compile-verified, matrix operand order checked by reading, not run on hardware
//
#include <hip/hip_runtime.h>
#include <math.h>
#include <stdint.h>

// ---------------------------------------------------------------------------
// MI455X / gfx1250 implementation.
// Core GEMMs use v_wmma_f32_16x16x32_f16 (wave32 WMMA), f16 operands staged
// through LDS, f32 accumulation, fused bias+GELU epilogue.
// Weights are pre-converted once per call to f16 [N][K] (transposed) so both
// GEMM tiles are contiguous 16B rows -> b128 LDS staging via the CDNA5 async
// global->LDS path (GLOBAL_LOAD_ASYNC_TO_LDS_B128 / ASYNCcnt) when available.
// Cross-attention is algebraically collapsed: it has exactly 1 KV token, so
// softmax == 1 and the cross-attn output is just vc broadcast over tokens.
// ---------------------------------------------------------------------------

typedef __attribute__((ext_vector_type(16))) _Float16 v16h;
typedef __attribute__((ext_vector_type(8)))  _Float16 v8h;
typedef __attribute__((ext_vector_type(8)))  float    v8f;

#define NB   2048      // batch
#define NJ   56        // tokens
#define ND   256       // model dim
#define NH   8         // heads
#define NL   4         // layers

// Async global->LDS copies (CDNA5). Guarded: falls back to load+store if the
// builtin is not declared by this toolchain.
#if defined(__has_builtin)
#if __has_builtin(__builtin_amdgcn_global_load_async_to_lds_b128) && \
    __has_builtin(__builtin_amdgcn_s_wait_asynccnt)
#define USE_ASYNC_LDS 1
#endif
#endif

// Builtin signature (from clang diagnostic): param0 is
// 'int __attribute__((vector_size(16))) __device__ *' (AS1), param1 LDS (AS3).
typedef int b128i __attribute__((vector_size(16)));
// Global pointer -> AS(1) int4*; LDS generic pointer -> 32-bit AS(3) offset
// (low 32 bits of the generic encoding are the LDS offset on AMDGPU).
#define AS1V4(p) ((__attribute__((address_space(1))) b128i*)(uintptr_t)(p))
#define AS3V4(p) ((__attribute__((address_space(3))) b128i*)(unsigned)(uintptr_t)(p))

__device__ __forceinline__ void lds_copy16(const _Float16* src, _Float16* dst) {
#if defined(USE_ASYNC_LDS)
    __builtin_amdgcn_global_load_async_to_lds_b128(AS1V4(src), AS3V4(dst), 0, 0);
#else
    *(uint4*)dst = *(const uint4*)src;
#endif
}

__device__ __forceinline__ float gelu_f(float x) {
    const float k0 = 0.7978845608028654f, k1 = 0.044715f;
    return 0.5f * x * (1.0f + tanhf(k0 * (x + k1 * x * x * x)));
}

// ---------------------------------------------------------------------------
// Weight prep: f32 [K][N] -> f16 [N][K] (transposed), done once per call.
// ---------------------------------------------------------------------------
__global__ void wtrans_kernel(const float* __restrict__ W,
                              _Float16* __restrict__ Wt, int K, int N)
{
    int idx = blockIdx.x * 256 + threadIdx.x;
    if (idx >= K * N) return;
    int n = idx / K, k = idx - n * K;
    Wt[(size_t)n * K + k] = (_Float16)W[(size_t)k * N + n];
}

// ---------------------------------------------------------------------------
// WMMA GEMM:  out[M x N] = act( A_f16[M x K] @ Wt_f16[N x K]^T + bias )
// Block tile 128x128, 8 waves, each wave computes a 32x64 patch (2x4 WMMAs
// per 32-deep K step -> 8 static v_wmma per iteration).
// A rows can be remapped: phys_row = (m/grp)*gstride + m%grp + goff (grp==0 ->
// identity); remap is K-invariant, computed once before the K loop (32-bit).
// Requires: M%128==0, N%128==0, K%32==0, lda%8==0 (all call sites satisfy).
// ---------------------------------------------------------------------------
__global__ __launch_bounds__(256) void gemm_wmma_kernel(
    const _Float16* __restrict__ A, const _Float16* __restrict__ Wt,
    const float* __restrict__ bias, float* __restrict__ out32,
    _Float16* __restrict__ out16, int M, int N, int K, int lda,
    int grp, int gstride, int goff, int do_gelu)
{
    __shared__ _Float16 lA[128][40];    // 128x32 A tile, +8 pad (80B row, 16B aligned)
    __shared__ _Float16 lB[128][40];    // 128x32 Wt tile (row n, col k)

    const int tid  = threadIdx.x;
    const int wave = tid >> 5;
    const int lane = tid & 31;
    const int l16  = lane & 15;
    const int sel  = lane >> 4;         // 0: K 0..7 & 16..23   1: K 8..15 & 24..31
    const int wm   = wave >> 1;         // 0..3 (32-row band)
    const int wn   = wave & 1;          // 0..1 (64-col half)
    const int blockM = blockIdx.y * 128;
    const int n0     = blockIdx.x * 128;

    // ---- K-invariant A-row remap, 32-bit only, hoisted out of the K loop ----
    unsigned prow[2];
    #pragma unroll
    for (int it = 0; it < 2; ++it) {
        unsigned r = (unsigned)(tid + (it << 8)) >> 2;       // 0..127
        unsigned m = (unsigned)blockM + r;
        prow[it] = grp ? ((m / (unsigned)grp) * (unsigned)gstride +
                          (m % (unsigned)grp) + (unsigned)goff)
                       : m;
    }
    const _Float16* Wbase = Wt + (size_t)n0 * K;

    v8f acc[2][4] = {};

    const int nk = K >> 5;
    for (int kb = 0; kb < nk; ++kb) {
        // ---- stage A tile + W tile (each 128x32 f16, 512 x 16B), 4/thread ----
        #pragma unroll
        for (int it = 0; it < 2; ++it) {
            int v  = tid + (it << 8);
            int r  = v >> 2;
            int c8 = (v & 3) << 3;
            const _Float16* srcA = A + (size_t)prow[it] * lda + (kb << 5) + c8;
            lds_copy16(srcA, &lA[r][c8]);
            const _Float16* srcW = Wbase + (size_t)r * K + (kb << 5) + c8;
            lds_copy16(srcW, &lB[r][c8]);
            if (kb + 1 < nk) {   // speculative prefetch of next K tile
                __builtin_prefetch(srcA + 32, 0, 0);
                __builtin_prefetch(srcW + 32, 0, 0);
            }
        }
#if defined(USE_ASYNC_LDS)
        __builtin_amdgcn_s_wait_asynccnt(0);
#endif
        __syncthreads();

        // ---- fragments per CDNA5 16-bit A/B layout (CDNA5 ISA §7.12.2) ----
        v16h af[2], bf[4];
        #pragma unroll
        for (int rr = 0; rr < 2; ++rr) {
            int rb = wm * 32 + rr * 16 + l16;
            v8h lo = *(const v8h*)&lA[rb][sel * 8];
            v8h hi = *(const v8h*)&lA[rb][16 + sel * 8];
            #pragma unroll
            for (int e = 0; e < 8; ++e) { af[rr][e] = lo[e]; af[rr][8 + e] = hi[e]; }
        }
        #pragma unroll
        for (int cc = 0; cc < 4; ++cc) {
            int cb = wn * 64 + cc * 16 + l16;
            v8h lo = *(const v8h*)&lB[cb][sel * 8];
            v8h hi = *(const v8h*)&lB[cb][16 + sel * 8];
            #pragma unroll
            for (int e = 0; e < 8; ++e) { bf[cc][e] = lo[e]; bf[cc][8 + e] = hi[e]; }
        }
        #pragma unroll
        for (int rr = 0; rr < 2; ++rr)
            #pragma unroll
            for (int cc = 0; cc < 4; ++cc)
                acc[rr][cc] = __builtin_amdgcn_wmma_f32_16x16x32_f16(
                    false, af[rr], false, bf[cc], (short)0, acc[rr][cc], false, false);
        __syncthreads();
    }

    // ---- epilogue: C/D layout VGPR i -> row sel*8+i, col = lane&15 ----
    #pragma unroll
    for (int rr = 0; rr < 2; ++rr) {
        #pragma unroll
        for (int cc = 0; cc < 4; ++cc) {
            int colg = n0 + wn * 64 + cc * 16 + l16;
            float bv = bias[colg];
            #pragma unroll
            for (int i = 0; i < 8; ++i) {
                int rowg = blockM + wm * 32 + rr * 16 + sel * 8 + i;
                float val = acc[rr][cc][i] + bv;
                if (do_gelu) val = gelu_f(val);
                size_t o = (size_t)rowg * N + colg;
                if (out32) out32[o] = val;
                if (out16) out16[o] = (_Float16)val;
            }
        }
    }
}

// ---------------------------------------------------------------------------
// Self-attention: one block per batch element, 8 waves; wave w owns rows
// {w, w+8, ...}. Dh=32 = wave32 lanes. Writes o (f16) and, on last layer,
// attn mean-over-heads to d_out.
// ---------------------------------------------------------------------------
__global__ __launch_bounds__(256) void attn_kernel(
    const _Float16* __restrict__ qkv, const float* __restrict__ rel,
    _Float16* __restrict__ o, float* __restrict__ attn_out, int write_attn)
{
    __shared__ _Float16 lq[NJ][32], lk[NJ][32], lv[NJ][32];
    __shared__ float lattn[NJ][NJ + 1];

    const int b = blockIdx.x, tid = threadIdx.x;
    const int w = tid >> 5, lane = tid & 31;
    const float scale = 0.17677669529663687f;   // 1/sqrt(32)

    if (write_attn)
        for (int i = tid; i < NJ * (NJ + 1); i += 256) (&lattn[0][0])[i] = 0.0f;

    for (int h = 0; h < NH; ++h) {
        __syncthreads();
        for (int i = tid; i < NJ * 32; i += 256) {
            int j = i >> 5, d = i & 31;
            size_t base = ((size_t)b * NJ + j) * 768 + h * 32 + d;
            lq[j][d] = qkv[base];
            lk[j][d] = qkv[base + 256];
            lv[j][d] = qkv[base + 512];
        }
        __syncthreads();

        for (int it = 0; it < 7; ++it) {
            int r  = it * 8 + w;
            int c1 = lane + 32;
            bool v1 = (c1 < NJ);
            float s0 = 0.f, s1 = 0.f;
            #pragma unroll
            for (int d = 0; d < 32; ++d) {
                float qd = (float)lq[r][d];
                s0 += qd * (float)lk[lane][d];
                if (v1) s1 += qd * (float)lk[c1][d];
            }
            s0 = s0 * scale + rel[r * NJ + lane];
            s1 = v1 ? (s1 * scale + rel[r * NJ + c1]) : -1e30f;

            float mx = fmaxf(s0, s1);
            for (int m = 16; m > 0; m >>= 1) mx = fmaxf(mx, __shfl_xor(mx, m, 32));
            float e0 = __expf(s0 - mx);
            float e1 = v1 ? __expf(s1 - mx) : 0.f;
            float sm = e0 + e1;
            for (int m = 16; m > 0; m >>= 1) sm += __shfl_xor(sm, m, 32);
            float a0 = e0 / sm, a1 = e1 / sm;

            // o[r][lane] = sum_c a[c]*v[c][lane]; a[c] gathered via shfl
            float oa = 0.f;
            #pragma unroll
            for (int c = 0; c < 32; ++c) oa += __shfl(a0, c, 32) * (float)lv[c][lane];
            #pragma unroll
            for (int c = 32; c < NJ; ++c) oa += __shfl(a1, c - 32, 32) * (float)lv[c][lane];
            o[((size_t)b * NJ + r) * ND + h * 32 + lane] = (_Float16)oa;

            if (write_attn) {
                lattn[r][lane] += a0 * 0.125f;
                if (v1) lattn[r][c1] += a1 * 0.125f;
            }
        }
    }
    __syncthreads();
    if (write_attn)
        for (int i = tid; i < NJ * NJ; i += 256)
            attn_out[(size_t)b * NJ * NJ + i] = lattn[i / NJ][i % NJ];
}

// ---------------------------------------------------------------------------
// Residual + LayerNorm over D=256 (one block per token, one thread per chan).
// y_per_batch: residual is a per-batch vector (collapsed cross-attention).
// Emits f32 x and mirrored f16 copy for the next WMMA GEMM.
// ---------------------------------------------------------------------------
__global__ __launch_bounds__(256) void resln_kernel(
    const float* __restrict__ x, const float* __restrict__ y, int y_per_batch,
    const float* __restrict__ g, const float* __restrict__ bt,
    float* __restrict__ xo, _Float16* __restrict__ xh)
{
    __shared__ float red[256];
    const int row = blockIdx.x, c = threadIdx.x;
    const int b = row / NJ;
    float v = x[(size_t)row * ND + c] +
              (y_per_batch ? y[(size_t)b * ND + c] : y[(size_t)row * ND + c]);
    red[c] = v; __syncthreads();
    for (int s = 128; s > 0; s >>= 1) { if (c < s) red[c] += red[c + s]; __syncthreads(); }
    float mean = red[0] * (1.f / 256.f); __syncthreads();
    float d = v - mean;
    red[c] = d * d; __syncthreads();
    for (int s = 128; s > 0; s >>= 1) { if (c < s) red[c] += red[c + s]; __syncthreads(); }
    float var = red[0] * (1.f / 256.f);
    float o = d * rsqrtf(var + 1e-5f) * g[c] + bt[c];
    xo[(size_t)row * ND + c] = o;
    xh[(size_t)row * ND + c] = (_Float16)o;
}

// ---------------------------------------------------------------------------
// Token embedding: pose MLP (6->256 gelu ->256) / transl_emb, mask select,
// + joints_emb, input LayerNorm. One block per token.
// ---------------------------------------------------------------------------
__global__ __launch_bounds__(256) void embed_kernel(
    const float* __restrict__ body_pose, const float* __restrict__ pW1,
    const float* __restrict__ pb1, const float* __restrict__ pW2,
    const float* __restrict__ pb2, const float* __restrict__ joints_emb,
    const float* __restrict__ mask_emb, const float* __restrict__ transl_emb,
    const float* __restrict__ in_g, const float* __restrict__ in_b,
    const int* __restrict__ mask_ids, float* __restrict__ xf,
    _Float16* __restrict__ xh)
{
    __shared__ float lh[256];
    __shared__ float red[256];
    const int row = blockIdx.x, c = threadIdx.x;
    const int b = row / NJ, j = row % NJ;       // block-uniform

    if (j < 55) {
        float h = pb1[c];
        #pragma unroll
        for (int k = 0; k < 6; ++k)
            h += body_pose[((size_t)b * 55 + j) * 6 + k] * pW1[k * ND + c];
        lh[c] = gelu_f(h);
    }
    __syncthreads();

    float x0;
    if (j < 55) {
        float a = pb2[c];
        for (int k = 0; k < 256; ++k) a += lh[k] * pW2[k * ND + c];
        x0 = a;
    } else {
        x0 = transl_emb[c];
    }
    if (!mask_ids[b * NJ + j]) x0 = mask_emb[c];
    x0 += joints_emb[j * ND + c];

    red[c] = x0; __syncthreads();
    for (int s = 128; s > 0; s >>= 1) { if (c < s) red[c] += red[c + s]; __syncthreads(); }
    float mean = red[0] * (1.f / 256.f); __syncthreads();
    float d = x0 - mean;
    red[c] = d * d; __syncthreads();
    for (int s = 128; s > 0; s >>= 1) { if (c < s) red[c] += red[c + s]; __syncthreads(); }
    float var = red[0] * (1.f / 256.f);
    float o = d * rsqrtf(var + 1e-5f) * in_g[c] + in_b[c];
    xf[(size_t)row * ND + c] = o;
    xh[(size_t)row * ND + c] = (_Float16)o;
}

// ---------------------------------------------------------------------------
// Condition MLP: concat[wrist(10), betas(3), gender_emb(16)] -> 512 gelu ->
// 256 -> LN. One block per batch element. Output f16 (feeds ca_Wkv GEMM).
// ---------------------------------------------------------------------------
__global__ __launch_bounds__(256) void cond_kernel(
    const float* __restrict__ wrist, const float* __restrict__ betas,
    const float* __restrict__ gtab, const int* __restrict__ gender,
    const float* __restrict__ W1, const float* __restrict__ b1,
    const float* __restrict__ W2, const float* __restrict__ b2,
    const float* __restrict__ g, const float* __restrict__ bb,
    _Float16* __restrict__ cond_h)
{
    __shared__ float lin[32];
    __shared__ float lh[512];
    __shared__ float red[256];
    const int b = blockIdx.x, tid = threadIdx.x;

    if (tid < 10)       lin[tid] = wrist[b * 10 + tid];
    else if (tid < 13)  lin[tid] = betas[b * 3 + tid - 10];
    else if (tid < 29)  lin[tid] = gtab[gender[b] * 16 + tid - 13];
    __syncthreads();

    for (int i = tid; i < 512; i += 256) {
        float h = b1[i];
        #pragma unroll
        for (int k = 0; k < 29; ++k) h += lin[k] * W1[k * 512 + i];
        lh[i] = gelu_f(h);
    }
    __syncthreads();

    float a = b2[tid];
    for (int k = 0; k < 512; ++k) a += lh[k] * W2[k * ND + tid];

    red[tid] = a; __syncthreads();
    for (int s = 128; s > 0; s >>= 1) { if (tid < s) red[tid] += red[tid + s]; __syncthreads(); }
    float mean = red[0] * (1.f / 256.f); __syncthreads();
    float d = a - mean;
    red[tid] = d * d; __syncthreads();
    for (int s = 128; s > 0; s >>= 1) { if (tid < s) red[tid] += red[tid + s]; __syncthreads(); }
    float var = red[0] * (1.f / 256.f);
    cond_h[(size_t)b * ND + tid] = (_Float16)(d * rsqrtf(var + 1e-5f) * g[tid] + bb[tid]);
}

// ---------------------------------------------------------------------------
// Output heads (tiny N): 1024 -> 6 and 1024 -> 1 projections.
// ---------------------------------------------------------------------------
__global__ void rot_out_kernel(const _Float16* __restrict__ roth,
                               const float* __restrict__ W2,
                               const float* __restrict__ b2,
                               float* __restrict__ out)
{
    int idx = blockIdx.x * 256 + threadIdx.x;
    if (idx >= 2048 * 55 * 6) return;
    int m = idx / 6, c = idx % 6;
    float acc = b2[c];
    const _Float16* r = roth + (size_t)m * 1024;
    for (int k = 0; k < 1024; ++k) acc += (float)r[k] * W2[k * 6 + c];
    out[idx] = acc;
}

__global__ void tr_out_kernel(const _Float16* __restrict__ trh,
                              const float* __restrict__ W2,
                              const float* __restrict__ b2,
                              float* __restrict__ out)
{
    int b = blockIdx.x * 256 + threadIdx.x;
    if (b >= NB) return;
    float acc = b2[0];
    const _Float16* r = trh + (size_t)b * 1024;
    for (int k = 0; k < 1024; ++k) acc += (float)r[k] * W2[k];
    out[b] = acc;
}

// ---------------------------------------------------------------------------
extern "C" void kernel_launch(void* const* d_in, const int* in_sizes, int n_in,
                              void* d_out, int out_size, void* d_ws, size_t ws_size,
                              hipStream_t stream)
{
    (void)in_sizes; (void)n_in; (void)out_size; (void)ws_size;

    const float* body_pose  = (const float*)d_in[0];
    const float* wrist      = (const float*)d_in[1];
    const float* betas      = (const float*)d_in[2];
    const float* pose_W1    = (const float*)d_in[3];
    const float* pose_b1    = (const float*)d_in[4];
    const float* pose_W2    = (const float*)d_in[5];
    const float* pose_b2    = (const float*)d_in[6];
    const float* joints_emb = (const float*)d_in[7];
    const float* mask_emb   = (const float*)d_in[8];
    const float* transl_emb = (const float*)d_in[9];
    const float* in_g       = (const float*)d_in[10];
    const float* in_b       = (const float*)d_in[11];
    const float* gender_tab = (const float*)d_in[12];
    const float* cond_W1    = (const float*)d_in[13];
    const float* cond_b1    = (const float*)d_in[14];
    const float* cond_W2    = (const float*)d_in[15];
    const float* cond_b2    = (const float*)d_in[16];
    const float* cond_g     = (const float*)d_in[17];
    const float* cond_bb    = (const float*)d_in[18];
    const float* rel_map    = (const float*)d_in[19];
    const float* sa_Wqkv    = (const float*)d_in[20];
    const float* sa_bqkv    = (const float*)d_in[21];
    const float* sa_Wo      = (const float*)d_in[22];
    const float* sa_bo      = (const float*)d_in[23];
    // d_in[24]/[25] (ca_Wq/ca_bq) provably don't affect outputs: softmax over
    // a single key is identically 1, so cross-attn output == vc broadcast.
    const float* ca_Wkv     = (const float*)d_in[26];
    const float* ca_bkv     = (const float*)d_in[27];
    const float* ca_Wo      = (const float*)d_in[28];
    const float* ca_bo      = (const float*)d_in[29];
    const float* ff_W1      = (const float*)d_in[30];
    const float* ff_b1      = (const float*)d_in[31];
    const float* ff_W2      = (const float*)d_in[32];
    const float* ff_b2      = (const float*)d_in[33];
    const float* ln1_g      = (const float*)d_in[34];
    const float* ln1_b      = (const float*)d_in[35];
    const float* ln2_g      = (const float*)d_in[36];
    const float* ln2_b      = (const float*)d_in[37];
    const float* ln3_g      = (const float*)d_in[38];
    const float* ln3_b      = (const float*)d_in[39];
    const float* rot_W1     = (const float*)d_in[40];
    const float* rot_b1     = (const float*)d_in[41];
    const float* rot_W2     = (const float*)d_in[42];
    const float* rot_b2     = (const float*)d_in[43];
    const float* tr_W1      = (const float*)d_in[44];
    const float* tr_b1      = (const float*)d_in[45];
    const float* tr_W2      = (const float*)d_in[46];
    const float* tr_b2      = (const float*)d_in[47];
    const int*   gender     = (const int*)d_in[48];
    const int*   mask_ids   = (const int*)d_in[49];

    const size_t BJ = (size_t)NB * NJ;     // 114688

    // workspace carve-out (aligned to 256B)
    char* ws = (char*)d_ws;
    size_t off = 0;
    auto carve = [&](size_t bytes) -> void* {
        void* p = ws + off;
        off = (off + bytes + 255) & ~(size_t)255;
        return p;
    };
    float*    xf    = (float*)   carve(BJ * ND * 4);          // activations f32
    _Float16* xh    = (_Float16*)carve(BJ * ND * 2);          // f16 mirror
    float*    y     = (float*)   carve(BJ * ND * 4);          // pre-LN GEMM out
    _Float16* bigh  = (_Float16*)carve(BJ * 1024 * 2);        // qkv / ff-hidden / rot-hidden
    _Float16* oh    = (_Float16*)carve(BJ * ND * 2);          // attn output
    _Float16* condh = (_Float16*)carve((size_t)NB * ND * 2);
    _Float16* kvh   = (_Float16*)carve((size_t)NB * 512 * 2);
    float*    co    = (float*)   carve((size_t)NB * ND * 4);  // collapsed cross-attn
    _Float16* trh   = (_Float16*)carve((size_t)NB * 1024 * 2);
    // pre-transposed f16 weights [N][K]
    _Float16* wt_qkv  = (_Float16*)carve((size_t)NL * 768  * ND * 2);
    _Float16* wt_wo   = (_Float16*)carve((size_t)NL * ND   * ND * 2);
    _Float16* wt_kv   = (_Float16*)carve((size_t)NL * 512  * ND * 2);
    _Float16* wt_cawo = (_Float16*)carve((size_t)NL * ND   * 512 * 2);
    _Float16* wt_ff1  = (_Float16*)carve((size_t)NL * 1024 * ND * 2);
    _Float16* wt_ff2  = (_Float16*)carve((size_t)NL * ND   * 1024 * 2);
    _Float16* wt_rot  = (_Float16*)carve((size_t)1024 * ND * 2);
    _Float16* wt_tr   = (_Float16*)carve((size_t)1024 * ND * 2);

    float* out_pose   = (float*)d_out;                  // (B,55,6)
    float* out_transl = out_pose + (size_t)NB * 55 * 6; // (B,1)
    float* out_attn   = out_transl + NB;                // (B,56,56)

    auto wtrans = [&](const float* W, _Float16* Wt, int K, int N) {
        wtrans_kernel<<<(K * N + 255) / 256, 256, 0, stream>>>(W, Wt, K, N);
    };
    auto gemm = [&](const _Float16* A, const _Float16* Wt, const float* bias,
                    float* o32, _Float16* o16, int M, int N, int K, int lda,
                    int grp, int gs, int go, int act) {
        dim3 grid(N / 128, M / 128);
        gemm_wmma_kernel<<<grid, 256, 0, stream>>>(A, Wt, bias, o32, o16, M, N, K,
                                                   lda, grp, gs, go, act);
    };

    // ---- one-time weight conversion/transposition (f32 [K][N] -> f16 [N][K]) ----
    for (int l = 0; l < NL; ++l) {
        wtrans(sa_Wqkv + (size_t)l * ND * 768,  wt_qkv  + (size_t)l * 768  * ND, ND, 768);
        wtrans(sa_Wo   + (size_t)l * ND * ND,   wt_wo   + (size_t)l * ND   * ND, ND, ND);
        wtrans(ca_Wkv  + (size_t)l * ND * 512,  wt_kv   + (size_t)l * 512  * ND, ND, 512);
        wtrans(ca_Wo   + (size_t)l * ND * ND,   wt_cawo + (size_t)l * ND   * 512, ND, ND);
        wtrans(ff_W1   + (size_t)l * ND * 1024, wt_ff1  + (size_t)l * 1024 * ND, ND, 1024);
        wtrans(ff_W2   + (size_t)l * 1024 * ND, wt_ff2  + (size_t)l * ND * 1024, 1024, ND);
    }
    wtrans(rot_W1, wt_rot, ND, 1024);
    wtrans(tr_W1,  wt_tr,  ND, 1024);

    // ---- input embedding + cond branch ----
    embed_kernel<<<(int)BJ, 256, 0, stream>>>(body_pose, pose_W1, pose_b1, pose_W2,
        pose_b2, joints_emb, mask_emb, transl_emb, in_g, in_b, mask_ids, xf, xh);
    cond_kernel<<<NB, 256, 0, stream>>>(wrist, betas, gender_tab, gender, cond_W1,
        cond_b1, cond_W2, cond_b2, cond_g, cond_bb, condh);

    // ---- transformer layers ----
    for (int l = 0; l < NL; ++l) {
        // self-attention
        gemm(xh, wt_qkv + (size_t)l * 768 * ND, sa_bqkv + (size_t)l * 768,
             nullptr, bigh, (int)BJ, 768, ND, ND, 0, 0, 0, 0);
        attn_kernel<<<NB, 256, 0, stream>>>(bigh, rel_map, oh, out_attn,
                                            (l == NL - 1) ? 1 : 0);
        gemm(oh, wt_wo + (size_t)l * ND * ND, sa_bo + (size_t)l * ND,
             y, nullptr, (int)BJ, ND, ND, ND, 0, 0, 0, 0);
        resln_kernel<<<(int)BJ, 256, 0, stream>>>(xf, y, 0,
            ln1_g + (size_t)l * ND, ln1_b + (size_t)l * ND, xf, xh);

        // collapsed cross-attention: co = vc @ Wo + bo, vc = (cond@Wkv+b)[:,D:]
        gemm(condh, wt_kv + (size_t)l * 512 * ND, ca_bkv + (size_t)l * 512,
             nullptr, kvh, NB, 512, ND, ND, 0, 0, 0, 0);
        gemm(kvh + ND, wt_cawo + (size_t)l * ND * 512, ca_bo + (size_t)l * ND,
             co, nullptr, NB, ND, ND, 512, 0, 0, 0, 0);
        resln_kernel<<<(int)BJ, 256, 0, stream>>>(xf, co, 1,
            ln2_g + (size_t)l * ND, ln2_b + (size_t)l * ND, xf, xh);

        // feed-forward
        gemm(xh, wt_ff1 + (size_t)l * 1024 * ND, ff_b1 + (size_t)l * 1024,
             nullptr, bigh, (int)BJ, 1024, ND, ND, 0, 0, 0, 1);
        gemm(bigh, wt_ff2 + (size_t)l * ND * 1024, ff_b2 + (size_t)l * ND,
             y, nullptr, (int)BJ, ND, 1024, 1024, 0, 0, 0, 0);
        resln_kernel<<<(int)BJ, 256, 0, stream>>>(xf, y, 0,
            ln3_g + (size_t)l * ND, ln3_b + (size_t)l * ND, xf, xh);
    }

    // ---- output heads ----
    // rot head on tokens j<55: A rows remapped (m/55)*56 + m%55
    gemm(xh, wt_rot, rot_b1, nullptr, bigh, NB * 55, 1024, ND, ND, 55, 56, 0, 1);
    rot_out_kernel<<<(NB * 55 * 6 + 255) / 256, 256, 0, stream>>>(bigh, rot_W2,
                                                                  rot_b2, out_pose);
    // transl head on token j==55: A rows remapped m*56 + 55
    gemm(xh, wt_tr, tr_b1, nullptr, trh, NB, 1024, ND, ND, 1, 56, 55, 1);
    tr_out_kernel<<<(NB + 255) / 256, 256, 0, stream>>>(trh, tr_W2, tr_b2,
                                                        out_transl);
}
